// CRF_1D_24352464569974
// MI455X (gfx1250) — matrix operand order repaired and checked by reading, not compiled, not run on hardware
//
#include <hip/hip_runtime.h>
#include <hip/hip_bf16.h>

// CRF forward algorithm on MI455X (gfx1250, wave32, WMMA).
//
// Chunked parallel scan over time: each wave owns one (batch, chunk) and
// builds the chunk's 32x32 transition operator in the scaled-exp domain via
// v_wmma_f32_16x16x32_f16 (K=32 == NUM_TAGS; 2x2 output tiles => 4 WMMAs per
// step). Per-COLUMN max-rescales (column scales commute through the matmul).
// Raw D values are f16-packed BEFORE the rescale, so the 2 column-max plus
// 8 packed-data exchanges are all independent; a single inline-asm block
// issues all 10 ds_bpermute_b32 back-to-back with ONE s_wait_dscnt -> one
// LDS round-trip per step on the recurrence critical path.

#define Bsz    512
#define Tlen   2048
#define KTAG   32
#define CHUNKS 16
#define CLEN   128   // steps per chunk (covers t = 1..2047)

typedef __attribute__((ext_vector_type(16))) _Float16 v16h;
typedef __attribute__((ext_vector_type(8)))  float    v8f;

#define LN2F 0.6931471805599453f

union VU { v16h v; int w[8]; };

__device__ __forceinline__ int pack16(float a, float b) {
  auto p = __builtin_amdgcn_cvt_pkrtz(a, b);  // saturates, no Inf
  int r;
  __builtin_memcpy(&r, &p, 4);
  return r;
}

__device__ __forceinline__ float wave_max(float v) {
  int lane = (int)(threadIdx.x & 31u);
  for (int xm = 16; xm >= 1; xm >>= 1) {
    int src = (lane ^ xm) << 2;
    float o = __int_as_float(__builtin_amdgcn_ds_bpermute(src, __float_as_int(v)));
    v = fmaxf(v, o);
  }
  return v;
}

__device__ __forceinline__ float wave_sum(float v) {
  int lane = (int)(threadIdx.x & 31u);
  for (int xm = 16; xm >= 1; xm >>= 1) {
    int src = (lane ^ xm) << 2;
    v += __int_as_float(__builtin_amdgcn_ds_bpermute(src, __float_as_int(v)));
  }
  return v;
}

// ---------------------------------------------------------------------------
// Phase 1: per (batch, chunk) build P_chunk = prod_t (diag(d_t) E^T) with
// per-step per-column rescale; store P-hat (f32, row j contiguous) and the
// per-column log-scales.
// ---------------------------------------------------------------------------
__global__ __launch_bounds__(256) void crf_phase1(
    const float* __restrict__ emissions, const unsigned char* __restrict__ mask,
    const float* __restrict__ trans, float* __restrict__ Pm,
    float* __restrict__ lscArr) {
  const int tid    = threadIdx.x;
  const int lane   = tid & 31;
  const int waveId = (int)((blockIdx.x * blockDim.x + tid) >> 5);
  const int b      = waveId / CHUNKS;
  const int c      = waveId % CHUNKS;
  if (b >= Bsz) return;

  const int  lj   = lane & 15;
  const bool hi   = lane >= 16;
  const int  xidx = (lane ^ 16) << 2;  // bpermute byte index, loop-invariant

  // Static A-layout tiles of E^T: EA[jt] row j = jt*16+lj; halves -> K index:
  // lanes 0-15 hold K 0..15; lanes 16-31 hold K 16..31.
  v16h EA0, EA1;
  #pragma unroll
  for (int h = 0; h < 16; ++h) {
    int k = hi ? (h + 16) : h;
    EA0[h] = (_Float16)__expf(trans[k * KTAG + lj]);
    EA1[h] = (_Float16)__expf(trans[k * KTAG + 16 + lj]);
  }

  // P-hat = identity in B-operand layout (lanes 0-15 hold K=0..15, lanes
  // 16-31 hold K=16..31; N = lane&15). Two N-tiles PB0/PB1.
  v16h PB0 = {}, PB1 = {};
  if (!hi) PB0[lj] = (_Float16)1.0f;
  else     PB1[lj] = (_Float16)1.0f;

  // D accumulators (C layout): element (jt,nt,v): row j = jt*16+v+8*hi,
  // col i = nt*16+lj. Identity init so a fully-masked chunk stores I.
  v8f D[2][2];
  #pragma unroll
  for (int jt = 0; jt < 2; ++jt)
    #pragma unroll
    for (int nt = 0; nt < 2; ++nt)
      #pragma unroll
      for (int v = 0; v < 8; ++v)
        D[jt][nt][v] = ((jt == nt) && ((v + (hi ? 8 : 0)) == lj)) ? 1.0f : 0.0f;

  float inv0 = 1.0f, inv1 = 1.0f;   // per-column scales (this lane's columns)
  float L0 = 0.0f, L1 = 0.0f;       // per-column log-scales
  const int t0 = 1 + c * CLEN;
  int steps = (Tlen - 1) - c * CLEN;
  if (steps > CLEN) steps = CLEN;

  const float*         eb = emissions + (size_t)b * Tlen * KTAG;
  const unsigned char* mb = mask + (size_t)b * Tlen;

  for (int s = 0; s < steps; ++s) {
    const int t = t0 + s;
    if (!mb[t]) continue;  // masked step => identity operator (wave-uniform)

    const float* et = eb + (size_t)t * KTAG;
    const _Float16 d0 = (_Float16)__expf(et[lj]);        // row scale j=lj
    const _Float16 d1 = (_Float16)__expf(et[16 + lj]);   // row scale j=16+lj

    const v16h A0 = EA0 * d0;   // ext-vector * scalar -> v_pk_mul_f16
    const v16h A1 = EA1 * d1;

    const v8f z = {};
    D[0][0] = __builtin_amdgcn_wmma_f32_16x16x32_f16(false, A0, false, PB0,
                                                     (short)0, z, false, false);
    D[0][1] = __builtin_amdgcn_wmma_f32_16x16x32_f16(false, A0, false, PB1,
                                                     (short)0, z, false, false);
    D[1][0] = __builtin_amdgcn_wmma_f32_16x16x32_f16(false, A1, false, PB0,
                                                     (short)0, z, false, false);
    D[1][1] = __builtin_amdgcn_wmma_f32_16x16x32_f16(false, A1, false, PB1,
                                                     (short)0, z, false, false);

    // (a) Balanced per-column max over this lane's 16 rows of each column.
    float mo[2];
    #pragma unroll
    for (int nt = 0; nt < 2; ++nt) {
      float a0 = fmaxf(D[0][nt][0], D[1][nt][0]);
      float a1 = fmaxf(D[0][nt][1], D[1][nt][1]);
      float a2 = fmaxf(D[0][nt][2], D[1][nt][2]);
      float a3 = fmaxf(D[0][nt][3], D[1][nt][3]);
      float a4 = fmaxf(D[0][nt][4], D[1][nt][4]);
      float a5 = fmaxf(D[0][nt][5], D[1][nt][5]);
      float a6 = fmaxf(D[0][nt][6], D[1][nt][6]);
      float a7 = fmaxf(D[0][nt][7], D[1][nt][7]);
      float b0 = fmaxf(fmaxf(a0, a1), fmaxf(a2, a3));
      float b1 = fmaxf(fmaxf(a4, a5), fmaxf(a6, a7));
      mo[nt] = fmaxf(b0, b1);
    }

    // (b) Pack RAW D into f16 pairs (bounded ~12k < 65504; rtz saturates)
    //     and pick what the partner needs: lo lanes send jt1, hi send jt0.
    int h0[2][4], h1[2][4], snd[2][4];
    #pragma unroll
    for (int nt = 0; nt < 2; ++nt)
      #pragma unroll
      for (int k = 0; k < 4; ++k) {
        h0[nt][k] = pack16(D[0][nt][2 * k], D[0][nt][2 * k + 1]);
        h1[nt][k] = pack16(D[1][nt][2 * k], D[1][nt][2 * k + 1]);
        snd[nt][k] = hi ? h0[nt][k] : h1[nt][k];
      }

    // (c) All 10 independent exchanges issued back-to-back with ONE wait.
    //     Early-clobber outputs: results land asynchronously before the
    //     trailing s_wait_dscnt, so outputs must not alias inputs.
    int mr0, mr1, r00, r01, r02, r03, r10, r11, r12, r13;
    asm volatile(
        "ds_bpermute_b32 %0, %10, %11\n\t"
        "ds_bpermute_b32 %1, %10, %12\n\t"
        "ds_bpermute_b32 %2, %10, %13\n\t"
        "ds_bpermute_b32 %3, %10, %14\n\t"
        "ds_bpermute_b32 %4, %10, %15\n\t"
        "ds_bpermute_b32 %5, %10, %16\n\t"
        "ds_bpermute_b32 %6, %10, %17\n\t"
        "ds_bpermute_b32 %7, %10, %18\n\t"
        "ds_bpermute_b32 %8, %10, %19\n\t"
        "ds_bpermute_b32 %9, %10, %20\n\t"
        "s_wait_dscnt 0x0"
        : "=&v"(mr0), "=&v"(mr1), "=&v"(r00), "=&v"(r01), "=&v"(r02),
          "=&v"(r03), "=&v"(r10), "=&v"(r11), "=&v"(r12), "=&v"(r13)
        : "v"(xidx), "v"(__float_as_int(mo[0])), "v"(__float_as_int(mo[1])),
          "v"(snd[0][0]), "v"(snd[0][1]), "v"(snd[0][2]), "v"(snd[0][3]),
          "v"(snd[1][0]), "v"(snd[1][1]), "v"(snd[1][2]), "v"(snd[1][3]));

    // (d) Finalize scales (same mx feeds rcp and the log accumulator).
    const float mx0 = fmaxf(mo[0], __int_as_float(mr0));
    const float mx1 = fmaxf(mo[1], __int_as_float(mr1));
    inv0 = __builtin_amdgcn_rcpf(mx0);
    inv1 = __builtin_amdgcn_rcpf(mx1);
    L0 += LN2F * __builtin_amdgcn_logf(mx0);   // v_log_f32 is log2
    L1 += LN2F * __builtin_amdgcn_logf(mx1);

    // (e) Assemble raw next-B operands, then apply the per-column scale as
    //     one uniform v_pk_mul_f16 per operand.
    const int rc0[4] = {r00, r01, r02, r03};
    const int rc1[4] = {r10, r11, r12, r13};
    VU r0, r1;
    #pragma unroll
    for (int k = 0; k < 4; ++k) {
      r0.w[k]     = hi ? rc0[k] : h0[0][k];  // rows 16*hi   + 0..7
      r0.w[4 + k] = hi ? h1[0][k] : rc0[k];  // rows 16*hi+8 + 0..7
      r1.w[k]     = hi ? rc1[k] : h0[1][k];
      r1.w[4 + k] = hi ? h1[1][k] : rc1[k];
    }
    PB0 = r0.v * (_Float16)inv0;
    PB1 = r1.v * (_Float16)inv1;
  }

  // Store P-hat[j][i] (f32, row-major): half-wave-coalesced 64B rows.
  float* Pout = Pm + (size_t)(b * CHUNKS + c) * (KTAG * KTAG);
  #pragma unroll
  for (int jt = 0; jt < 2; ++jt)
    #pragma unroll
    for (int nt = 0; nt < 2; ++nt)
      #pragma unroll
      for (int v = 0; v < 8; ++v) {
        const int j = jt * 16 + v + (hi ? 8 : 0);
        const int i = nt * 16 + lj;
        Pout[j * KTAG + i] = D[jt][nt][v] * (nt ? inv1 : inv0);
      }
  // Per-column log-scales (lanes l and l^16 hold identical values).
  if (!hi) {
    float* Lout = lscArr + (size_t)(b * CHUNKS + c) * KTAG;
    Lout[lj]      = L0;
    Lout[16 + lj] = L1;
  }
}

// ---------------------------------------------------------------------------
// Numerator: gathers along tags + block reduction. Runs after phase 1 so the
// whole emissions tensor is resident in the 192MB L2.
// ---------------------------------------------------------------------------
__global__ __launch_bounds__(256) void crf_numerator(
    const float* __restrict__ emissions, const int* __restrict__ tags,
    const unsigned char* __restrict__ mask, const float* __restrict__ trans,
    const float* __restrict__ start_trans, const float* __restrict__ end_trans,
    float* __restrict__ numOut) {
  __shared__ float sAcc[256];
  __shared__ int   sCnt[256];
  const int b = blockIdx.x, tid = threadIdx.x;
  const int*           tg = tags + (size_t)b * Tlen;
  const unsigned char* mb = mask + (size_t)b * Tlen;
  const float*         eb = emissions + (size_t)b * Tlen * KTAG;

  float acc = 0.0f;
  int   cnt = 0;
  for (int t = tid; t < Tlen; t += 256) {
    const float m   = mb[t] ? 1.0f : 0.0f;
    const int   tag = tg[t];
    acc += m * eb[(size_t)t * KTAG + tag];
    if (t >= 1) acc += m * trans[tg[t - 1] * KTAG + tag];
    cnt += mb[t] ? 1 : 0;
  }
  sAcc[tid] = acc;
  sCnt[tid] = cnt;
  __syncthreads();
  for (int o = 128; o >= 1; o >>= 1) {
    if (tid < o) { sAcc[tid] += sAcc[tid + o]; sCnt[tid] += sCnt[tid + o]; }
    __syncthreads();
  }
  if (tid == 0) {
    const int last = sCnt[0] - 1;
    numOut[b] = sAcc[0] + start_trans[tg[0]] + end_trans[tg[last]];
  }
}

// ---------------------------------------------------------------------------
// Phase 2: one wave per batch; thread alpha through the 16 chunk operators.
// Per-column log-scales fold into alpha before each matvec (exact).
// ---------------------------------------------------------------------------
__global__ __launch_bounds__(256) void crf_phase2(
    const float* __restrict__ emissions, const float* __restrict__ start_trans,
    const float* __restrict__ end_trans, const float* __restrict__ Pm,
    const float* __restrict__ lscArr, const float* __restrict__ numArr,
    float* __restrict__ out) {
  const int lane = threadIdx.x & 31;
  const int b    = (int)(blockIdx.x * (blockDim.x >> 5) + (threadIdx.x >> 5));
  if (b >= Bsz) return;
  const int j = lane;

  float alpha = start_trans[j] + emissions[(size_t)b * Tlen * KTAG + j];

  for (int c = 0; c < CHUNKS; ++c) {
    const float aeff = alpha + lscArr[(size_t)(b * CHUNKS + c) * KTAG + j];
    const float m    = wave_max(aeff);
    const float ae   = __expf(aeff - m);
    const int   ai   = __float_as_int(ae);
    const float* Prow = Pm + (size_t)(b * CHUNKS + c) * (KTAG * KTAG) + j * KTAG;
    float s = 0.0f;
    #pragma unroll
    for (int i = 0; i < KTAG; ++i) {
      const float aei = __int_as_float(__builtin_amdgcn_readlane(ai, i));
      s = fmaf(aei, Prow[i], s);
    }
    alpha = m + __logf(s);
  }

  alpha += end_trans[j];
  const float m2   = wave_max(alpha);
  const float se   = wave_sum(__expf(alpha - m2));
  const float logz = m2 + __logf(se);
  if (lane == 0) out[b] = logz - numArr[b];
}

// ---------------------------------------------------------------------------
extern "C" void kernel_launch(void* const* d_in, const int* in_sizes, int n_in,
                              void* d_out, int out_size, void* d_ws,
                              size_t ws_size, hipStream_t stream) {
  const float*         emissions   = (const float*)d_in[0];
  const int*           tags        = (const int*)d_in[1];
  const unsigned char* mask        = (const unsigned char*)d_in[2];
  const float*         trans       = (const float*)d_in[3];
  const float*         start_trans = (const float*)d_in[4];
  const float*         end_trans   = (const float*)d_in[5];
  float*               out         = (float*)d_out;

  char*  ws  = (char*)d_ws;
  float* Pm  = (float*)ws;                                   // 32 MiB
  float* lsc = (float*)(ws + (size_t)Bsz * CHUNKS * KTAG * KTAG * sizeof(float));
  float* num = lsc + (size_t)Bsz * CHUNKS * KTAG;            // 1 MiB of lsc

  // Phase 1: 512*16 = 8192 waves, 8 waves/block.
  crf_phase1<<<(Bsz * CHUNKS) / 8, 256, 0, stream>>>(emissions, mask, trans,
                                                     Pm, lsc);
  // Numerator: one block per batch.
  crf_numerator<<<Bsz, 256, 0, stream>>>(emissions, tags, mask, trans,
                                         start_trans, end_trans, num);
  // Phase 2: one wave per batch.
  crf_phase2<<<Bsz / 8, 256, 0, stream>>>(emissions, start_trans, end_trans,
                                          Pm, lsc, num, out);
}
